// MoEMLP_49520972923489
// MI455X (gfx1250) — matrix operand order repaired and checked by reading, not compiled, not run on hardware
//
#include <hip/hip_runtime.h>
#include <hip/hip_bf16.h>

// ---------------- problem constants ----------------
constexpr int DM   = 1024;   // d_model
constexpr int DH   = 4096;   // d_hidden
constexpr int NE   = 8;      // experts
constexpr int NTOK = 8192;   // B*T
constexpr int CAP  = 1281;   // int(1.25*(N/E)+1)

constexpr int BM = 128, BN = 64, BK = 32;
constexpr int LDP = BK + 8;  // LDS row pitch in halves (40 -> 80B, 16B multiple)

typedef __attribute__((ext_vector_type(16))) __bf16 bf16x16;
typedef __attribute__((ext_vector_type(8)))  float  f32x8;
typedef __attribute__((ext_vector_type(4)))  float  f32x4;
typedef __attribute__((ext_vector_type(2)))  float  f32x2;
typedef __attribute__((ext_vector_type(4)))  unsigned int u32x4;
typedef __attribute__((ext_vector_type(2)))  unsigned int u32x2;

union ABFrag { bf16x16 v; u32x4 q[2]; };
union Pack8  { __bf16 h[8]; u32x4 q; };
union Pack4  { __bf16 h[4]; u32x2 q; };

__device__ inline float gelu_tanh(float x) {
    float x3 = x * x * x;
    float t  = tanhf(0.7978845608028654f * (x + 0.044715f * x3));
    return 0.5f * x * (1.0f + t);
}

// ---------------- kernel 1: zero output ----------------
__global__ void zero_out_kernel(float* __restrict__ out, int n) {
    int i = blockIdx.x * 256 + threadIdx.x;
    if (i < n) out[i] = 0.0f;
}

// ---------------- kernel 2: init map / loss sums ----------------
__global__ void init_misc_kernel(int* __restrict__ map, float* __restrict__ sums) {
    int i = blockIdx.x * 256 + threadIdx.x;
    if (i < NE * CAP) map[i] = -1;
    if (i < 16) sums[i] = 0.0f;
}

// ---------------- kernel 3: router (one wave32 per token) ----------------
__global__ __launch_bounds__(256) void router_kernel(
    const float* __restrict__ x, const float* __restrict__ Wr,
    const float* __restrict__ br,
    int* __restrict__ top1, float* __restrict__ wgt, float* __restrict__ sums)
{
    __shared__ float sprob[NE];
    __shared__ float scnt[NE];
    int tid = threadIdx.x;
    if (tid < NE) { sprob[tid] = 0.0f; scnt[tid] = 0.0f; }
    __syncthreads();

    int lane  = tid & 31;
    int wave  = tid >> 5;
    int token = blockIdx.x * 8 + wave;

    float p[NE];
#pragma unroll
    for (int e = 0; e < NE; ++e) p[e] = 0.0f;

    const float* xr = x + (size_t)token * DM;
    for (int d = lane; d < DM; d += 32) {
        float xv = xr[d];
        const float* wr = Wr + d * NE;
#pragma unroll
        for (int e = 0; e < NE; ++e) p[e] += xv * wr[e];
    }
#pragma unroll
    for (int off = 16; off > 0; off >>= 1) {
#pragma unroll
        for (int e = 0; e < NE; ++e) p[e] += __shfl_xor(p[e], off, 32);
    }

    if (lane == 0) {
        float lg[NE], mx = -3.4e38f;
#pragma unroll
        for (int e = 0; e < NE; ++e) { lg[e] = p[e] + br[e]; mx = fmaxf(mx, lg[e]); }
        float pe[NE], s = 0.0f;
#pragma unroll
        for (int e = 0; e < NE; ++e) { pe[e] = __expf(lg[e] - mx); s += pe[e]; }
        float inv = 1.0f / s;
        int am = 0; float best = pe[0];
#pragma unroll
        for (int e = 1; e < NE; ++e) if (pe[e] > best) { best = pe[e]; am = e; }
        top1[token] = am;
        wgt[token]  = best * inv;
#pragma unroll
        for (int e = 0; e < NE; ++e) atomicAdd(&sprob[e], pe[e] * inv);
        atomicAdd(&scnt[am], 1.0f);
    }
    __syncthreads();
    if (tid < NE) {
        atomicAdd(&sums[tid],      sprob[tid]);
        atomicAdd(&sums[NE + tid], scnt[tid]);
    }
}

// ---------------- kernel 4: token-order dispatch scan ----------------
__global__ __launch_bounds__(1024) void scan_kernel(
    const int* __restrict__ top1, int* __restrict__ map)
{
    __shared__ int cnt[1024][NE];
    int t = threadIdx.x;
    int myE[8];
    int loc[NE];
#pragma unroll
    for (int e = 0; e < NE; ++e) loc[e] = 0;
#pragma unroll
    for (int i = 0; i < 8; ++i) {
        int e = top1[t * 8 + i];
        myE[i] = e;
        loc[e] += 1;
    }
#pragma unroll
    for (int e = 0; e < NE; ++e) cnt[t][e] = loc[e];
    __syncthreads();

    for (int off = 1; off < 1024; off <<= 1) {
        int v[NE];
        bool act = (t >= off);
        if (act) {
#pragma unroll
            for (int e = 0; e < NE; ++e) v[e] = cnt[t - off][e];
        }
        __syncthreads();
        if (act) {
#pragma unroll
            for (int e = 0; e < NE; ++e) cnt[t][e] += v[e];
        }
        __syncthreads();
    }

    int run[NE];
#pragma unroll
    for (int e = 0; e < NE; ++e) run[e] = (t > 0) ? cnt[t - 1][e] : 0;
#pragma unroll
    for (int i = 0; i < 8; ++i) {
        int e = myE[i];
        int pos = run[e]++;
        if (pos < CAP) map[e * CAP + pos] = t * 8 + i;
    }
}

// ---------------- kernel 5: FC1 (gather + GEMM + GELU) ----------------
__global__ __launch_bounds__(256) void fc1_kernel(
    const float* __restrict__ x, const float* __restrict__ W1,
    const float* __restrict__ b1, const int* __restrict__ map,
    __bf16* __restrict__ he)
{
    constexpr int tilesM = (CAP + BM - 1) / BM;   // 11
    constexpr int tilesN = DH / BN;               // 64
    int bid = blockIdx.x;
    int e   = bid / (tilesM * tilesN);
    int r   = bid % (tilesM * tilesN);
    int tm  = r / tilesN, tn = r % tilesN;
    int m0  = tm * BM, n0 = tn * BN;

    __shared__ __bf16 sA[BM][LDP];
    __shared__ __bf16 sB[BN][LDP];   // transposed: [col][k]

    int tid  = threadIdx.x;
    int lane = tid & 31;
    int wave = tid >> 5;
    int wm   = wave & 3;     // M sub-tile (32 rows)
    int wn   = wave >> 2;    // N sub-tile (32 cols)

    // ---- A loader mapping: 128 rows x 32 cols, 2 threads/row, 16 floats each
    int aRow  = tid >> 1;
    int aCol0 = (tid & 1) * 16;
    int gm    = m0 + aRow;
    int tokenA = (gm < CAP) ? map[e * CAP + gm] : -1;
    float amask = (tokenA >= 0) ? 1.0f : 0.0f;
    const float* xrow = x + (size_t)(tokenA < 0 ? 0 : tokenA) * DM;

    // ---- B loader mapping: 32 k x 64 n, thread = 4k x 2n micro-tile
    int kg = tid >> 5;        // 0..7  -> k rows kg*4..kg*4+3
    int np = tid & 31;        // 0..31 -> cols np*2, np*2+1
    const float* w1base = W1 + (size_t)e * DM * DH;

    f32x8 acc[2][2] = {};
    f32x4 aR[4];
    f32x2 bR[4];

    // prologue: load k-slice 0 into registers
#pragma unroll
    for (int i = 0; i < 4; ++i)
        aR[i] = *(const f32x4*)(xrow + aCol0 + i * 4);
#pragma unroll
    for (int j = 0; j < 4; ++j)
        bR[j] = *(const f32x2*)(w1base + (size_t)(kg * 4 + j) * DH + n0 + np * 2);

    for (int kt = 0; kt < DM / BK; ++kt) {
        // stage registers -> LDS (packed bf16, vector stores)
        Pack8 pa0, pa1;
#pragma unroll
        for (int i = 0; i < 4; ++i) {
            pa0.h[i]     = (__bf16)(aR[0][i] * amask);
            pa0.h[4 + i] = (__bf16)(aR[1][i] * amask);
            pa1.h[i]     = (__bf16)(aR[2][i] * amask);
            pa1.h[4 + i] = (__bf16)(aR[3][i] * amask);
        }
        *(u32x4*)&sA[aRow][aCol0]     = pa0.q;
        *(u32x4*)&sA[aRow][aCol0 + 8] = pa1.q;

        Pack4 pb0, pb1;
#pragma unroll
        for (int j = 0; j < 4; ++j) {
            pb0.h[j] = (__bf16)bR[j][0];
            pb1.h[j] = (__bf16)bR[j][1];
        }
        *(u32x2*)&sB[np * 2][kg * 4]     = pb0.q;
        *(u32x2*)&sB[np * 2 + 1][kg * 4] = pb1.q;
        __syncthreads();

        // software pipeline: issue next k-slice global loads now
        if (kt + 1 < DM / BK) {
            int kN = (kt + 1) * BK;
#pragma unroll
            for (int i = 0; i < 4; ++i)
                aR[i] = *(const f32x4*)(xrow + kN + aCol0 + i * 4);
#pragma unroll
            for (int j = 0; j < 4; ++j)
                bR[j] = *(const f32x2*)(w1base + (size_t)(kN + kg * 4 + j) * DH + n0 + np * 2);
        }

        // fragments + 4 WMMA per wave
        int kb  = (lane >> 4) * 8;
        int kb2 = (lane >> 4) * 16;
        ABFrag fa[2], fb[2];
#pragma unroll
        for (int i = 0; i < 2; ++i) {
            int rA = wm * 32 + i * 16 + (lane & 15);
            fa[i].q[0] = *(const u32x4*)&sA[rA][kb];
            fa[i].q[1] = *(const u32x4*)&sA[rA][kb + 16];
        }
#pragma unroll
        for (int j = 0; j < 2; ++j) {
            int c = wn * 32 + j * 16 + (lane & 15);
            fb[j].q[0] = *(const u32x4*)&sB[c][kb];
            fb[j].q[1] = *(const u32x4*)&sB[c][kb + 16];
        }
#pragma unroll
        for (int i = 0; i < 2; ++i)
#pragma unroll
            for (int j = 0; j < 2; ++j)
                acc[i][j] = __builtin_amdgcn_wmma_f32_16x16x32_bf16(
                    false, fa[i].v, false, fb[j].v, (short)0, acc[i][j], false, false);
        __syncthreads();
    }

    // epilogue: bias + GELU -> he (bf16)
    int nLane = lane & 15;
    int mAdd  = (lane < 16) ? 0 : 8;
#pragma unroll
    for (int j = 0; j < 2; ++j) {
        int n = n0 + wn * 32 + j * 16 + nLane;
        float bias = b1[e * DH + n];
#pragma unroll
        for (int i = 0; i < 2; ++i) {
#pragma unroll
            for (int rr = 0; rr < 8; ++rr) {
                int m = m0 + wm * 32 + i * 16 + mAdd + rr;
                if (m < CAP) {
                    he[(size_t)(e * CAP + m) * DH + n] =
                        (__bf16)gelu_tanh(acc[i][j][rr] + bias);
                }
            }
        }
    }
}

// ---------------- kernel 6: FC2 (GEMM + weighted scatter-combine) ----------------
__global__ __launch_bounds__(256) void fc2_kernel(
    const __bf16* __restrict__ he, const float* __restrict__ W2,
    const float* __restrict__ b2, const int* __restrict__ map,
    const float* __restrict__ wgt, float* __restrict__ out)
{
    constexpr int tilesM = (CAP + BM - 1) / BM;   // 11
    constexpr int tilesN = DM / BN;               // 16
    int bid = blockIdx.x;
    int e   = bid / (tilesM * tilesN);
    int r   = bid % (tilesM * tilesN);
    int tm  = r / tilesN, tn = r % tilesN;
    int m0  = tm * BM, n0 = tn * BN;

    __shared__ __bf16 sA[BM][LDP];
    __shared__ __bf16 sB[BN][LDP];   // transposed: [col][k]

    int tid  = threadIdx.x;
    int lane = tid & 31;
    int wave = tid >> 5;
    int wm   = wave & 3;
    int wn   = wave >> 2;

    int aRow  = tid >> 1;
    int aCol0 = (tid & 1) * 16;
    int gm    = m0 + aRow;
    bool aval = (gm < CAP);
    const __bf16* herow = he + (size_t)(e * CAP + (aval ? gm : 0)) * DH;

    int kg = tid >> 5;
    int np = tid & 31;
    const float* w2base = W2 + (size_t)e * DH * DM;

    const u32x4 zq = {0u, 0u, 0u, 0u};
    f32x8 acc[2][2] = {};
    u32x4 hR[2];
    f32x2 bR[4];

#pragma unroll
    for (int i = 0; i < 2; ++i)
        hR[i] = *(const u32x4*)(herow + aCol0 + i * 8);
#pragma unroll
    for (int j = 0; j < 4; ++j)
        bR[j] = *(const f32x2*)(w2base + (size_t)(kg * 4 + j) * DM + n0 + np * 2);

    for (int kt = 0; kt < DH / BK; ++kt) {
        *(u32x4*)&sA[aRow][aCol0]     = aval ? hR[0] : zq;
        *(u32x4*)&sA[aRow][aCol0 + 8] = aval ? hR[1] : zq;

        Pack4 pb0, pb1;
#pragma unroll
        for (int j = 0; j < 4; ++j) {
            pb0.h[j] = (__bf16)bR[j][0];
            pb1.h[j] = (__bf16)bR[j][1];
        }
        *(u32x2*)&sB[np * 2][kg * 4]     = pb0.q;
        *(u32x2*)&sB[np * 2 + 1][kg * 4] = pb1.q;
        __syncthreads();

        if (kt + 1 < DH / BK) {
            int kN = (kt + 1) * BK;
#pragma unroll
            for (int i = 0; i < 2; ++i)
                hR[i] = *(const u32x4*)(herow + kN + aCol0 + i * 8);
#pragma unroll
            for (int j = 0; j < 4; ++j)
                bR[j] = *(const f32x2*)(w2base + (size_t)(kN + kg * 4 + j) * DM + n0 + np * 2);
        }

        int kb  = (lane >> 4) * 8;
        int kb2 = (lane >> 4) * 16;
        ABFrag fa[2], fb[2];
#pragma unroll
        for (int i = 0; i < 2; ++i) {
            int rA = wm * 32 + i * 16 + (lane & 15);
            fa[i].q[0] = *(const u32x4*)&sA[rA][kb];
            fa[i].q[1] = *(const u32x4*)&sA[rA][kb + 16];
        }
#pragma unroll
        for (int j = 0; j < 2; ++j) {
            int c = wn * 32 + j * 16 + (lane & 15);
            fb[j].q[0] = *(const u32x4*)&sB[c][kb];
            fb[j].q[1] = *(const u32x4*)&sB[c][kb + 16];
        }
#pragma unroll
        for (int i = 0; i < 2; ++i)
#pragma unroll
            for (int j = 0; j < 2; ++j)
                acc[i][j] = __builtin_amdgcn_wmma_f32_16x16x32_bf16(
                    false, fa[i].v, false, fb[j].v, (short)0, acc[i][j], false, false);
        __syncthreads();
    }

    // epilogue: bias, router weight, scatter to token rows
    int nLane = lane & 15;
    int mAdd  = (lane < 16) ? 0 : 8;
    float bias[2];
#pragma unroll
    for (int j = 0; j < 2; ++j)
        bias[j] = b2[e * DM + n0 + wn * 32 + j * 16 + nLane];
#pragma unroll
    for (int i = 0; i < 2; ++i) {
#pragma unroll
        for (int rr = 0; rr < 8; ++rr) {
            int m = m0 + wm * 32 + i * 16 + mAdd + rr;
            if (m >= CAP) continue;
            int token = map[e * CAP + m];
            if (token < 0) continue;
            float wk = wgt[token];
#pragma unroll
            for (int j = 0; j < 2; ++j) {
                int n = n0 + wn * 32 + j * 16 + nLane;
                out[(size_t)token * DM + n] = (acc[i][j][rr] + bias[j]) * wk;
            }
        }
    }
}

// ---------------- kernel 7: aux loss ----------------
__global__ void aux_kernel(const float* __restrict__ sums, float* __restrict__ auxout) {
    if (threadIdx.x == 0 && blockIdx.x == 0) {
        float lb = 0.0f;
#pragma unroll
        for (int e = 0; e < NE; ++e)
            lb += (sums[e] / (float)NTOK) * (sums[NE + e] / (float)NTOK);
        auxout[0] = 0.01f * ((float)NE * lb);
    }
}

// ---------------- host launcher ----------------
extern "C" void kernel_launch(void* const* d_in, const int* in_sizes, int n_in,
                              void* d_out, int out_size, void* d_ws, size_t ws_size,
                              hipStream_t stream) {
    const float* x  = (const float*)d_in[0];
    const float* Wr = (const float*)d_in[1];
    const float* br = (const float*)d_in[2];
    const float* W1 = (const float*)d_in[3];
    const float* b1 = (const float*)d_in[4];
    const float* W2 = (const float*)d_in[5];
    const float* b2 = (const float*)d_in[6];
    float* out = (float*)d_out;

    char* ws = (char*)d_ws;
    int*    top1 = (int*)(ws + 0);                       // 8192 * 4
    float*  wgt  = (float*)(ws + 32768);                 // 8192 * 4
    int*    map  = (int*)(ws + 65536);                   // NE*CAP*4 = 40992
    float*  sums = (float*)(ws + 65536 + 40992);         // 16 floats
    __bf16* he   = (__bf16*)(ws + 65536 + 40992 + 64);   // NE*CAP*DH bf16 (~84 MB)

    (void)in_sizes; (void)n_in; (void)ws_size;

    // 1. zero y + aux
    {
        int n = NTOK * DM + 1;
        zero_out_kernel<<<(n + 255) / 256, 256, 0, stream>>>(out, n);
    }
    // 2. init map + sums
    {
        int n = NE * CAP;
        init_misc_kernel<<<(n + 255) / 256, 256, 0, stream>>>(map, sums);
    }
    // 3. router
    router_kernel<<<NTOK / 8, 256, 0, stream>>>(x, Wr, br, top1, wgt, sums);
    // 4. dispatch scan
    scan_kernel<<<1, 1024, 0, stream>>>(top1, map);
    // 5. FC1
    {
        constexpr int tilesM = (CAP + BM - 1) / BM;
        constexpr int tilesN = DH / BN;
        fc1_kernel<<<NE * tilesM * tilesN, 256, 0, stream>>>(x, W1, b1, map, he);
    }
    // 6. FC2 + combine
    {
        constexpr int tilesM = (CAP + BM - 1) / BM;
        constexpr int tilesN = DM / BN;
        fc2_kernel<<<NE * tilesM * tilesN, 256, 0, stream>>>(he, W2, b2, map, wgt, out);
    }
    // 7. aux loss scalar
    aux_kernel<<<1, 64, 0, stream>>>(sums, out + (size_t)NTOK * DM);
}
